// Decoder_31052613550393
// MI455X (gfx1250) — compile-verified, matrix operand order checked by reading, not run on hardware
//
#include <hip/hip_runtime.h>
#include <hip/hip_bf16.h>
#include <cstdint>
#include <type_traits>

// ---------------------------------------------------------------------------
// Decoder (attention + GRU + vocab projection) for MI455X / gfx1250, wave32.
//   * hoist Uk = henc@Ua^T, gi_x = emb[x]@Wih_x^T + bih, s0 out of the scan
//   * serial loop over T=64: q-GEMM -> attn -> c-GEMM -> GRU gates
//   * final 2048x32000x1024 GEMM with v_wmma_f32_16x16x32_f16
// f16 operand tiles are staged LDS-side by the Tensor Data Mover
// (tensor_load_to_lds + s_wait_tensorcnt); f32 operands are converted to f16
// while staging (v_cvt_pk_f16_f32 + ds_store_b128). Fragment loads are all
// ds_load_b128 thanks to K-contiguous LDS tiles (A: As[m][k], B: Bs[n][k]).
// ---------------------------------------------------------------------------

typedef __attribute__((ext_vector_type(16))) _Float16 v16h;
typedef __attribute__((ext_vector_type(8)))  _Float16 v8h;
typedef __attribute__((ext_vector_type(8)))  float    v8f;
typedef __attribute__((ext_vector_type(4)))  unsigned int v4u;
typedef __attribute__((ext_vector_type(8)))  int v8i;
typedef __attribute__((ext_vector_type(4)))  int v4i;

#if defined(__has_builtin)
#if __has_builtin(__builtin_amdgcn_tensor_load_to_lds) && \
    __has_builtin(__builtin_amdgcn_s_wait_tensorcnt)
#define HAVE_TDM 1
#endif
#endif
#ifndef HAVE_TDM
#define HAVE_TDM 0
#endif

namespace cfg {
constexpr int B  = 32;
constexpr int T  = 64;
constexpr int Tx = 64;
constexpr int H  = 1024;
constexpr int E  = 512;
constexpr int V  = 32000;
constexpr int H2 = 2 * H;     // 2048
constexpr int H3 = 3 * H;     // 3072
constexpr int EH = E + H2;    // 2560 (Wih inner dim)
}

// ---------------------------------------------------------------------------
// TDM: async 2D f16 tile load (rows x 64 halves), global -> LDS, with LDS
// row padding of 16B after each 128B row (matches the 72-half LDS pitch).
// Issued by one wave; caller must s_wait_tensorcnt + barrier before use.
// Descriptor packing per CDNA5 ISA D# group0/group1 tables.
// ---------------------------------------------------------------------------
#if HAVE_TDM
__device__ __forceinline__ void tdm_load_tile_f16(
    unsigned int lds_byte_off, const _Float16* gsrc,
    int tile_rows, int row_stride_elems, int tensor_k, int tensor_rows)
{
    const unsigned long long ga = (unsigned long long)(uintptr_t)gsrc;
    v4u g0;
    g0.x = 1u;                                       // count=1 (user descriptor)
    g0.y = lds_byte_off;                             // lds_addr (bytes)
    g0.z = (unsigned int)(ga & 0xFFFFFFFFu);         // global_addr[31:0]
    g0.w = (unsigned int)((ga >> 32) & 0x01FFFFFFu)  // global_addr[56:32]
         | (2u << 30);                               // type = 2 ("image")
    v8i g1;
    g1[0] = (1 << 16)      // data_size = 2 bytes
          | (1 << 20)      // pad_enable
          | (4 << 22)      // pad_interval: 32 DWORDs (one 128B row)
          | (3 << 25);     // pad_amount: 4 DWORDs (16B)
    g1[1] = (tensor_k & 0xFFFF) << 16;                       // tensor_dim0[15:0]
    g1[2] = ((tensor_k >> 16) & 0xFFFF)                      // tensor_dim0[31:16]
          | ((tensor_rows & 0xFFFF) << 16);                  // tensor_dim1[15:0]
    g1[3] = ((tensor_rows >> 16) & 0xFFFF)                   // tensor_dim1[31:16]
          | (64 << 16);                                      // tile_dim0 = 64
    g1[4] = (tile_rows & 0xFFFF);                            // tile_dim1 ; tile_dim2=0
    g1[5] = row_stride_elems;                                // dim0_stride[31:0]
    g1[6] = 0;                                               // dim0_stride[47:32]
    g1[7] = 0;
    const v4i gz = {0, 0, 0, 0};
#if defined(__clang_major__) && (__clang_major__ >= 23)
    const v8i gz8 = {0, 0, 0, 0, 0, 0, 0, 0};
    __builtin_amdgcn_tensor_load_to_lds(g0, g1, gz, gz, gz8, 0);
#else
    __builtin_amdgcn_tensor_load_to_lds(g0, g1, gz, gz, 0);
#endif
}
#endif

// ---------------------------------------------------------------------------
// Generic WMMA GEMM:  C[M,N] (f32) = A[M,K] * op(B) + bias
//   BTRANS=true : B stored [N,K] row-major (computes A * B^T)  -- "y = x W^T"
//   BTRANS=false: B stored [K,N] row-major (transposed while staging to LDS)
// AT/BT may be float (converted to f16 while staging) or _Float16 (tile is
// DMA'd by the Tensor Data Mover).
// Block = 256 threads = 8 waves. BN=128, KT=64 (two WMMA k-steps per stage).
//   BM=64: waves 4(M) x 2(N), each wave 16x64 (4 accum tiles, A-frag reuse x4)
//   BM=32: waves 2(M) x 4(N), each wave 16x32 (2 accum tiles)
// Requires M%BM==0, N%128==0, K%64==0 (true for all shapes used here).
// ---------------------------------------------------------------------------
template <int BM, typename AT, typename BT, bool BTRANS>
__global__ __launch_bounds__(256)
void gemm_wmma(const AT* __restrict__ A, int lda,
               const BT* __restrict__ B, int ldb,
               const float* __restrict__ bias,
               float* __restrict__ C, int ldc, int K, int Mtot, int Ntot)
{
    static_assert(BM == 32 || BM == 64, "BM must be 32 or 64");
    constexpr int BN = 128;
    constexpr int KT = 64;
    constexpr int KP = KT + 8;              // 72-half pitch (144 B, 16B-aligned)
    constexpr int WAVES_M = BM / 16;        // 2 or 4
    constexpr int WAVES_N = 8 / WAVES_M;    // 4 or 2
    constexpr int WN = BN / WAVES_N;        // 32 or 64
    constexpr int NT = WN / 16;             // 2 or 4

    constexpr bool TDM_A = HAVE_TDM && std::is_same<AT, _Float16>::value;
    constexpr bool TDM_B = HAVE_TDM && BTRANS && std::is_same<BT, _Float16>::value;

    // Both tiles K-contiguous per row so fragment loads are b128.
    __shared__ __align__(16) _Float16 As[BM][KP];   // As[m][k]
    __shared__ __align__(16) _Float16 Bs[BN][KP];   // Bs[n][k] (== op(B)^T)

    const int tid  = threadIdx.x;
    const int lane = tid & 31;
    const int wv   = tid >> 5;
    const int wm   = wv % WAVES_M;
    const int wn   = wv / WAVES_M;
    const int hl   = lane >> 4;     // half-wave select (K-group)
    const int l16  = lane & 15;

    const int m0 = blockIdx.y * BM;
    const int n0 = blockIdx.x * BN;

    v8f acc[NT] = {};

    // A staging (manual path): APT contiguous K-elements per thread
    constexpr int APT   = BM * KT / 256;    // 8 (BM=32) or 16 (BM=64)
    constexpr int A_TPR = KT / APT;         // threads per A row
    const int arow  = tid / A_TPR;
    const int akoff = (tid % A_TPR) * APT;

    constexpr int BPT = KT * BN / 256;      // 32
    // BTRANS source [N,K]: straight copy, K-contiguous both sides
    const int bn    = tid / (KT / BPT);
    const int bkoff = (tid % (KT / BPT)) * BPT;
    // !BTRANS source [K,N]: transpose during staging
    const int bk    = tid / (BN / BPT);
    const int bnoff = (tid % (BN / BPT)) * BPT;

#if HAVE_TDM
    // LDS aperture keeps the byte offset in addr[31:0] (ISA aperture rules).
    const unsigned int AsOff = (unsigned int)(uintptr_t)&As[0][0];
    const unsigned int BsOff = (unsigned int)(uintptr_t)&Bs[0][0];
#endif

    for (int k0 = 0; k0 < K; k0 += KT) {
#if HAVE_TDM
        if constexpr (TDM_A) {
            if (wv == 0)
                tdm_load_tile_f16(AsOff,
                                  (const _Float16*)A + (size_t)m0 * lda + k0,
                                  BM, lda, K - k0, Mtot - m0);
        }
        if constexpr (TDM_B) {
            if (wv == 1)
                tdm_load_tile_f16(BsOff,
                                  (const _Float16*)B + (size_t)n0 * ldb + k0,
                                  BN, ldb, K - k0, Ntot - n0);
        }
#endif
        if constexpr (!TDM_A) {
#pragma unroll
            for (int e = 0; e < APT; ++e)
                As[arow][akoff + e] =
                    (_Float16)A[(size_t)(m0 + arow) * lda + (k0 + akoff + e)];
        }
        if constexpr (!TDM_B) {
            if constexpr (BTRANS) {
#pragma unroll
                for (int e = 0; e < BPT; ++e)
                    Bs[bn][bkoff + e] =
                        (_Float16)B[(size_t)(n0 + bn) * ldb + (k0 + bkoff + e)];
            } else {
#pragma unroll
                for (int e = 0; e < BPT; ++e)
                    Bs[bnoff + e][bk] =
                        (_Float16)B[(size_t)(k0 + bk) * ldb + (n0 + bnoff + e)];
            }
        }
#if HAVE_TDM
        if constexpr (TDM_A || TDM_B) {
            if (wv < 2) __builtin_amdgcn_s_wait_tensorcnt(0);
        }
#endif
        __syncthreads();

#pragma unroll
        for (int kk = 0; kk < KT; kk += 32) {
            // A fragment (16x32): lanes 0-15 hold K{0-7,16-23} of row m,
            // lanes 16-31 hold K{8-15,24-31}  -> two b128 LDS loads.
            const v8h alo = *(const v8h*)&As[wm * 16 + l16][kk + hl * 8];
            const v8h ahi = *(const v8h*)&As[wm * 16 + l16][kk + 16 + hl * 8];
            const v16h a = __builtin_shufflevector(alo, ahi,
                0, 1, 2, 3, 4, 5, 6, 7, 8, 9, 10, 11, 12, 13, 14, 15);
#pragma unroll
            for (int nt = 0; nt < NT; ++nt) {
                const int nc = wn * WN + nt * 16 + l16;
                const v8h blo = *(const v8h*)&Bs[nc][kk + hl * 8];
                const v8h bhi = *(const v8h*)&Bs[nc][kk + 16 + hl * 8];
                const v16h bf = __builtin_shufflevector(blo, bhi,
                    0, 1, 2, 3, 4, 5, 6, 7, 8, 9, 10, 11, 12, 13, 14, 15);
                acc[nt] = __builtin_amdgcn_wmma_f32_16x16x32_f16(
                    false, a, false, bf, (short)0, acc[nt], false, false);
            }
        }
        __syncthreads();
    }

    // C/D layout: lanes 0-15 -> M=i, lanes 16-31 -> M=8+i, N = lane&15
    const int lmBase = (lane < 16) ? 0 : 8;
#pragma unroll
    for (int nt = 0; nt < NT; ++nt) {
        const int gn = n0 + wn * WN + nt * 16 + l16;
        const float bv = bias ? bias[gn] : 0.0f;
#pragma unroll
        for (int i = 0; i < 8; ++i) {
            const int gm = m0 + wm * 16 + lmBase + i;
            C[(size_t)gm * ldc + gn] = acc[nt][i] + bv;
        }
    }
}

// ---------------------------------------------------------------------------
// Token embedding gather -> f16  ([B,T,E])
// ---------------------------------------------------------------------------
__global__ void embed_f16(const int* __restrict__ x, const float* __restrict__ emb,
                          _Float16* __restrict__ xe)
{
    const size_t idx = (size_t)blockIdx.x * 256 + threadIdx.x; // B*T*E
    const int e = (int)(idx % cfg::E);
    const size_t bt = idx / cfg::E;
    const int tok = x[bt];
    xe[idx] = (_Float16)emb[(size_t)tok * cfg::E + e];
}

// ---------------------------------------------------------------------------
// f32 -> f16 converters (full array / strided column slice)
// ---------------------------------------------------------------------------
__global__ void cvt_f16(const float* __restrict__ s, _Float16* __restrict__ d, int n)
{
    const int i = blockIdx.x * 256 + threadIdx.x;
    if (i < n) d[i] = (_Float16)s[i];
}

__global__ void cvt_submat_f16(const float* __restrict__ s, _Float16* __restrict__ d,
                               int rows, int cols, int ld, int c0)
{
    const int i = blockIdx.x * 256 + threadIdx.x;
    if (i < rows * cols) {
        const int r = i / cols, c = i % cols;
        d[i] = (_Float16)s[(size_t)r * ld + c0 + c];
    }
}

// ---------------------------------------------------------------------------
// Attention step (one block per batch element, 256 threads = 8 waves):
//   scores[tx] = sum_h tanh(q[b,h] + Uk[b,tx,h]) * Va[h]
//   w = softmax(scores);  c[b,:] = sum_tx w[tx] * henc[b,tx,:]
// ---------------------------------------------------------------------------
__global__ __launch_bounds__(256)
void attn_step(const float* __restrict__ q, const float* __restrict__ Uk,
               const float* __restrict__ Va, const float* __restrict__ henc,
               float* __restrict__ cF, _Float16* __restrict__ cH)
{
    using namespace cfg;
    const int b = blockIdx.x;
    const int tid = threadIdx.x;
    const int lane = tid & 31;
    const int wv = tid >> 5;

    __shared__ float sc[Tx];

    const float* qq = q + (size_t)b * H;
    for (int tx = wv; tx < Tx; tx += 8) {
        const float* uk = Uk + ((size_t)b * Tx + tx) * H;
        float p = 0.f;
        for (int h = lane; h < H; h += 32)
            p += tanhf(qq[h] + uk[h]) * Va[h];
#pragma unroll
        for (int off = 16; off > 0; off >>= 1)
            p += __shfl_xor(p, off, 32);
        if (lane == 0) sc[tx] = p;
    }
    __syncthreads();

    float mx = -3.0e38f;
    for (int i = 0; i < Tx; ++i) mx = fmaxf(mx, sc[i]);
    __syncthreads();
    if (tid < Tx) sc[tid] = expf(sc[tid] - mx);
    __syncthreads();
    float sum = 0.f;
    for (int i = 0; i < Tx; ++i) sum += sc[i];
    const float inv = 1.0f / sum;

    const float* hb = henc + (size_t)b * Tx * H2;
    for (int col = tid; col < H2; col += 256) {
        float acc = 0.f;
#pragma unroll 8
        for (int tx = 0; tx < Tx; ++tx)
            acc += sc[tx] * hb[(size_t)tx * H2 + col];
        acc *= inv;
        cF[(size_t)b * H2 + col] = acc;
        cH[(size_t)b * H2 + col] = (_Float16)acc;
    }
}

// ---------------------------------------------------------------------------
// GRU gates (h_prev == 0 in reference => hh-path contributes only bhh):
//   gi = gi_x[b,t,:] + gi_c[b,:]          (bih already folded into gi_x)
//   r = sig(gi_r + bhh_r); z = sig(gi_z + bhh_z); n = tanh(gi_n + r*bhh_n)
//   h = (1-z)*n     -> s (f32 + f16), Hdec[b,t,:] (f16)
// ---------------------------------------------------------------------------
__global__ void gru_gates(const float* __restrict__ giX, const float* __restrict__ giC,
                          const float* __restrict__ bhh, int t,
                          float* __restrict__ s, _Float16* __restrict__ sH,
                          _Float16* __restrict__ Hdec)
{
    using namespace cfg;
    const int idx = blockIdx.x * 256 + threadIdx.x;   // B*H = 32768
    const int b = idx >> 10;
    const int j = idx & (H - 1);
    const size_t bx = ((size_t)(b * T + t)) * H3;
    const size_t bc = (size_t)b * H3;

    const float gr = giX[bx + j]           + giC[bc + j];
    const float gz = giX[bx + H + j]       + giC[bc + H + j];
    const float gn = giX[bx + 2 * H + j]   + giC[bc + 2 * H + j];

    const float r = 1.f / (1.f + expf(-(gr + bhh[j])));
    const float z = 1.f / (1.f + expf(-(gz + bhh[H + j])));
    const float n = tanhf(gn + r * bhh[2 * H + j]);
    const float h = (1.f - z) * n;

    s[(size_t)b * H + j] = h;
    sH[(size_t)b * H + j] = (_Float16)h;
    Hdec[((size_t)(b * T + t)) * H + j] = (_Float16)h;
}

// ---------------------------------------------------------------------------
// Host launcher
// ---------------------------------------------------------------------------
extern "C" void kernel_launch(void* const* d_in, const int* in_sizes, int n_in,
                              void* d_out, int out_size, void* d_ws, size_t ws_size,
                              hipStream_t stream)
{
    using namespace cfg;
    (void)in_sizes; (void)n_in; (void)out_size; (void)ws_size;

    const int*   x    = (const int*)  d_in[0];   // [B,T]
    const float* henc = (const float*)d_in[1];   // [B,Tx,2H]
    const float* emb  = (const float*)d_in[2];   // [V,E]
    const float* WaW  = (const float*)d_in[3];   // [H,H]
    const float* WaB  = (const float*)d_in[4];   // [H]
    const float* UaW  = (const float*)d_in[5];   // [H,2H]
    const float* UaB  = (const float*)d_in[6];   // [H]
    const float* VaW  = (const float*)d_in[7];   // [1,H]
    const float* iniW = (const float*)d_in[8];   // [H,H]
    const float* Wih  = (const float*)d_in[9];   // [3H,E+2H]
    const float* bih  = (const float*)d_in[10];  // [3H]
    /* d_in[11] = Whh : unused (h_prev == 0)  */
    const float* bhh  = (const float*)d_in[12];  // [3H]
    const float* Wout = (const float*)d_in[13];  // [V,H]
    const float* bout = (const float*)d_in[14];  // [V]
    float* out = (float*)d_out;                  // [B,T,V]

    // ---- workspace bump allocator (~54 MB total) ----
    char* ws = (char*)d_ws;
    auto alloc = [&](size_t bytes) -> char* {
        char* p = ws;
        ws += (bytes + 255) & ~(size_t)255;
        return p;
    };
    _Float16* xe    = (_Float16*)alloc(sizeof(_Float16) * (size_t)B * T * E);   // [B,T,E]
    _Float16* WaH   = (_Float16*)alloc(sizeof(_Float16) * (size_t)H * H);       // [H,H]
    _Float16* WihcH = (_Float16*)alloc(sizeof(_Float16) * (size_t)H3 * H2);     // [3H,2H]
    float*    Uk    = (float*)   alloc(sizeof(float) * (size_t)B * Tx * H);     // [B,Tx,H]
    float*    giX   = (float*)   alloc(sizeof(float) * (size_t)B * T * H3);     // [B,T,3H]
    _Float16* Hdec  = (_Float16*)alloc(sizeof(_Float16) * (size_t)B * T * H);   // [B,T,H]
    float*    sF    = (float*)   alloc(sizeof(float) * (size_t)B * H);
    _Float16* sH    = (_Float16*)alloc(sizeof(_Float16) * (size_t)B * H);
    float*    qF    = (float*)   alloc(sizeof(float) * (size_t)B * H);
    float*    cF    = (float*)   alloc(sizeof(float) * (size_t)B * H2);
    _Float16* cH    = (_Float16*)alloc(sizeof(_Float16) * (size_t)B * H2);
    float*    giC   = (float*)   alloc(sizeof(float) * (size_t)B * H3);

    // ---- precompute (timestep-invariant) ----
    embed_f16<<<(B * T * E) / 256, 256, 0, stream>>>(x, emb, xe);
    cvt_f16<<<(H * H + 255) / 256, 256, 0, stream>>>(WaW, WaH, H * H);
    cvt_submat_f16<<<(H3 * H2 + 255) / 256, 256, 0, stream>>>(Wih, WihcH, H3, H2, EH, E);

    // Uk = henc @ Ua^T + Ua_b              [2048,1024] = [2048,2048]x[1024,2048]^T
    gemm_wmma<64, float, float, true>
        <<<dim3(H / 128, (B * Tx) / 64), 256, 0, stream>>>(
            henc, H2, UaW, H2, UaB, Uk, H, H2, B * Tx, H);

    // gi_x = xe @ Wih[:, :E]^T + bih       [2048,3072] = [2048,512]x[3072,2560(:512)]^T
    gemm_wmma<64, _Float16, float, true>
        <<<dim3(H3 / 128, (B * T) / 64), 256, 0, stream>>>(
            xe, E, Wih, EH, bih, giX, H3, E, B * T, H3);

    // s0 = henc[:,0,H:] @ initialW         [32,1024] = [32,1024]x[1024,1024] (B is [K,N])
    gemm_wmma<32, float, float, false>
        <<<dim3(H / 128, 1), 256, 0, stream>>>(
            henc + H, Tx * H2, iniW, H, nullptr, sF, H, H, B, H);
    cvt_f16<<<(B * H + 255) / 256, 256, 0, stream>>>(sF, sH, B * H);

    // ---- serial scan over T (s_t depends on s_{t-1}) ----
    for (int t = 0; t < T; ++t) {
        // q = s @ Wa^T + Wa_b              [32,1024]
        gemm_wmma<32, _Float16, _Float16, true>
            <<<dim3(H / 128, 1), 256, 0, stream>>>(sH, H, WaH, H, WaB, qF, H, H, B, H);

        // softmax(tanh(q+Uk)·Va) and context c  [32,2048]
        attn_step<<<B, 256, 0, stream>>>(qF, Uk, VaW, henc, cF, cH);

        // gi_c = c @ Wih[:, E:]^T          [32,3072]
        gemm_wmma<32, _Float16, _Float16, true>
            <<<dim3(H3 / 128, 1), 256, 0, stream>>>(cH, H2, WihcH, H2, nullptr, giC, H3, H2, B, H3);

        // gates -> h -> s, Hdec[:,t,:]
        gru_gates<<<(B * H) / 256, 256, 0, stream>>>(giX, giC, bhh, t, sF, sH, Hdec);
    }

    // logits = Hdec @ Wout^T + bout        [2048,32000] = [2048,1024]x[32000,1024]^T
    gemm_wmma<64, _Float16, float, true>
        <<<dim3(V / 128, (B * T) / 64), 256, 0, stream>>>(
            Hdec, H, Wout, H, bout, out, V, H, B * T, V);
}
